// PointPairwiseAttention_43078521979325
// MI455X (gfx1250) — matrix operand order repaired and checked by reading, not compiled, not run on hardware
//
#include <hip/hip_runtime.h>

typedef __attribute__((ext_vector_type(16))) __bf16 v16bf;
typedef __attribute__((ext_vector_type(8)))  float  v8f;

// Sizes fixed by the reference
#define BATCH 2
#define NPTS  1024
#define NK    1024
#define C16   16
#define CPW   32
#define HH    4
#define CO16  16

#define LOG2E 1.4426950408889634f

// Raw v_exp_f32 (base-2), no subnormal guard: weights < 2^-126 flush to 0,
// which is exactly the right behavior inside a softmax accumulation.
#define EXP2_RAW(x) __builtin_amdgcn_exp2f(x)

// ---------------------------------------------------------------------------
// Phase 0: per-point projections (rank factorization of the pairwise MLP
// first layers):  x_pw @ W = xq@(W_top - W_bot) + xk@W_bot
//   qw[row,32] = xq@(w1_top - w1_bot) + b1      kw[row,32] = xk@w1_bot
//   qv[row,16] = xq@(v1_top - v1_bot) + bv1     kv[row,16] = xk@v1_bot
// ---------------------------------------------------------------------------
__global__ __launch_bounds__(32)
void ppa_proj_kernel(const float* __restrict__ xq, const float* __restrict__ xk,
                     const float* __restrict__ w1, const float* __restrict__ b1,
                     const float* __restrict__ v1, const float* __restrict__ bv1,
                     float* __restrict__ qw, float* __restrict__ kw,
                     float* __restrict__ qv, float* __restrict__ kv)
{
    const int row = blockIdx.x;          // 0 .. B*N-1
    const int j   = threadIdx.x;         // 0 .. 31
    const float* xqr = xq + row * C16;
    const float* xkr = xk + row * C16;

    float aq = 0.f, ak = 0.f, aqv = 0.f, akv = 0.f;
#pragma unroll
    for (int c = 0; c < C16; ++c) {
        const float xqc = xqr[c];
        const float xkc = xkr[c];
        const float w1t = w1[c * CPW + j];
        const float w1b = w1[(C16 + c) * CPW + j];
        aq += xqc * (w1t - w1b);
        ak += xkc * w1b;
        if (j < C16) {
            const float v1t = v1[c * CO16 + j];
            const float v1b = v1[(C16 + c) * CO16 + j];
            aqv += xqc * (v1t - v1b);
            akv += xkc * v1b;
        }
    }
    qw[row * CPW + j] = aq + b1[j];
    kw[row * CPW + j] = ak;
    if (j < C16) {
        qv[row * C16 + j] = aqv + bv1[j];
        kv[row * C16 + j] = akv;
    }
}

// ---------------------------------------------------------------------------
// Phase 1: one wave per point n.  For each 16-row k tile:
//   logits tile  = relu(qw+kw)[16x32]bf16 @ (w2*log2e, cols replicated 4x)  (1 WMMA)
//   value  tile  = relu(qv+kv)[16x16->pad32]bf16 @ v2pad[32x16]             (1 WMMA)
// Column replication makes lane co's own D column the h = co>>2 logit column
// for exactly its own k rows -> no cross-lane shuffles in the hot loop.
// log2(e) folded into w2/b2 -> softmax runs in log2 domain on bare v_exp_f32.
// ---------------------------------------------------------------------------
#define WPB 8          // waves per block
#define KC  64         // k rows staged in LDS per outer iteration
#define KWS 36         // padded LDS row stride for kw (bank-conflict free)
#define KVS 20         // padded LDS row stride for kv

__global__ __launch_bounds__(WPB * 32)
void ppa_attn_kernel(const float* __restrict__ qw, const float* __restrict__ kw,
                     const float* __restrict__ qv, const float* __restrict__ kv,
                     const float* __restrict__ w2, const float* __restrict__ b2,
                     const float* __restrict__ v2, const float* __restrict__ bv2,
                     float* __restrict__ out)
{
    __shared__ float kwS[KC * KWS];
    __shared__ float kvS[KC * KVS];

    const int tid  = threadIdx.x;
    const int lane = tid & 31;
    const int wid  = tid >> 5;
    const int nIdx = blockIdx.x * WPB + wid;      // 0 .. B*N-1 (block stays in one b)
    const int bb   = nIdx >> 10;

    const int co   = lane & 15;                   // output column this lane owns (C/D layout)
    const int hh   = co >> 2;                     // softmax head this lane belongs to
    const int half = lane >> 4;                   // which 16-lane half
    const int jo0  = half ? 8 : 0;                // A-frag K sub-ranges per documented layout
    const int jo1  = jo0 + 16;

    // ---- B fragments (constant): lane holds column N=co; K pairs per VGPR.
    //      elements 0-7: K=8*half+e, elements 8-15: K=16+8*half+(e-8) (matches A fragment order)
    v16bf bW2, bV2;
    {
        const int kbase = half ? 16 : 0;
#pragma unroll
        for (int e = 0; e < 16; ++e) {
            const int k = (e < 8) ? (jo0 + e) : (jo1 + e - 8);
            const float w2v = w2[k * HH + hh] * LOG2E;            // replicate col 4x, log2e fold
            const float v2v = (k < C16) ? v2[k * CO16 + co] : 0.f; // v2 zero-padded K 16->32
            bW2[e] = (__bf16)w2v;
            bV2[e] = (__bf16)v2v;
        }
    }

    // ---- per-wave q-side rows (fixed n), loaded in A-fragment element order
    float qwr[16], qvr[8];
    {
        const float* q = qw + nIdx * CPW;
#pragma unroll
        for (int i = 0; i < 8; ++i) { qwr[i] = q[jo0 + i]; qwr[8 + i] = q[jo1 + i]; }
        const float* qvp = qv + nIdx * C16;
#pragma unroll
        for (int i = 0; i < 8; ++i) qvr[i] = qvp[jo0 + i];
    }
    const float b2c = b2[hh] * LOG2E;             // log2-domain bias
    const float bvc = bv2[co];

    float m = -1e30f, d = 0.f, acc = 0.f;         // online softmax state (log2 domain)

    const float* kwB = kw + (size_t)bb * NK * CPW;
    const float* kvB = kv + (size_t)bb * NK * C16;

    for (int k0 = 0; k0 < NK; k0 += KC) {
        // ---- cooperative, coalesced staging of 64 k rows into padded LDS
#pragma unroll
        for (int i = 0; i < 8; ++i) {
            const int idx = tid + i * 256;        // 0..2047
            const int r = idx >> 5, c = idx & 31;
            kwS[r * KWS + c] = kwB[(k0 + r) * CPW + c];
        }
#pragma unroll
        for (int i = 0; i < 4; ++i) {
            const int idx = tid + i * 256;        // 0..1023
            const int r = idx >> 4, c = idx & 15;
            kvS[r * KVS + c] = kvB[(k0 + r) * C16 + c];
        }
        __syncthreads();

        if (k0 + KC < NK)                          // pull next chunk toward L2/L0
            __builtin_prefetch(&kwB[(k0 + KC) * CPW + tid * 8], 0, 0);

#pragma unroll
        for (int sc = 0; sc < KC; sc += 16) {
            const int   krow = sc + co;            // A row M = lane&15 (documented layout)
            const float* kr  = &kwS[krow * KWS];
            const float* vr  = &kvS[krow * KVS];

            // ---- build bf16 A fragments with the fused add + ReLU
            v16bf a, av;
#pragma unroll
            for (int e = 0; e < 8; ++e) {
                a[e]      = (__bf16)fmaxf(qwr[e]     + kr[jo0 + e], 0.f);
                a[8 + e]  = (__bf16)fmaxf(qwr[8 + e] + kr[jo1 + e], 0.f);
                av[e]     = (__bf16)fmaxf(qvr[e]     + vr[jo0 + e], 0.f);
                av[8 + e] = (__bf16)0.f;           // K padding 16 -> 32
            }

            v8f c0 = {};
            v8f dlog = __builtin_amdgcn_wmma_f32_16x16x32_bf16(
                false, a,  false, bW2, (short)0, c0, false, false);
            v8f dval = __builtin_amdgcn_wmma_f32_16x16x32_bf16(
                false, av, false, bV2, (short)0, c0, false, false);

            // ---- bias + ReLU epilogue; lane already owns its h logit column
            float lg[8], val[8];
#pragma unroll
            for (int r = 0; r < 8; ++r) {
                lg[r]  = fmaxf(dlog[r] + b2c, 0.f);   // log2-domain logits (relu commutes, a>0)
                val[r] = fmaxf(dval[r] + bvc, 0.f);
            }

            // ---- online softmax update (bare v_exp_f32, no range guard)
            float mn = m;
#pragma unroll
            for (int r = 0; r < 8; ++r) mn = fmaxf(mn, lg[r]);
            const float scale = EXP2_RAW(m - mn);
            d *= scale; acc *= scale;
#pragma unroll
            for (int r = 0; r < 8; ++r) {
                const float w = EXP2_RAW(lg[r] - mn);
                d   += w;
                acc += w * val[r];
            }
            m = mn;
        }
        __syncthreads();
    }

    // ---- merge the two half-wave softmax states (k halves)
    const float mo = __shfl_xor(m,   16, 32);
    const float dd = __shfl_xor(d,   16, 32);
    const float ao = __shfl_xor(acc, 16, 32);
    const float mm = fmaxf(m, mo);
    const float s1 = EXP2_RAW(m - mm), s2 = EXP2_RAW(mo - mm);
    const float D  = d * s1 + dd * s2;
    const float A  = acc * s1 + ao * s2;
    if (half == 0)
        out[nIdx * CO16 + co] = A / D;
}

// ---------------------------------------------------------------------------
extern "C" void kernel_launch(void* const* d_in, const int* in_sizes, int n_in,
                              void* d_out, int out_size, void* d_ws, size_t ws_size,
                              hipStream_t stream)
{
    const float* xq  = (const float*)d_in[0];
    const float* xk  = (const float*)d_in[1];
    const float* w1  = (const float*)d_in[2];
    const float* b1  = (const float*)d_in[3];
    const float* w2  = (const float*)d_in[4];
    const float* b2  = (const float*)d_in[5];
    const float* v1  = (const float*)d_in[6];
    const float* bv1 = (const float*)d_in[7];
    const float* v2  = (const float*)d_in[8];
    const float* bv2 = (const float*)d_in[9];
    float* out = (float*)d_out;

    float* ws = (float*)d_ws;
    float* qw = ws;                                  // [2048,32]
    float* kw = ws + BATCH * NPTS * CPW;             // [2048,32]
    float* qv = kw + BATCH * NK * CPW;               // [2048,16]
    float* kv = qv + BATCH * NPTS * C16;             // [2048,16]

    ppa_proj_kernel<<<BATCH * NPTS, 32, 0, stream>>>(xq, xk, w1, b1, v1, bv1,
                                                     qw, kw, qv, kv);
    ppa_attn_kernel<<<(BATCH * NPTS) / WPB, WPB * 32, 0, stream>>>(
        qw, kw, qv, kv, w2, b2, v2, bv2, out);
}
